// AttentionPoolingModule_64415919505526
// MI455X (gfx1250) — compile-verified, matrix operand order checked by reading, not run on hardware
//
#include <hip/hip_runtime.h>
#include <hip/hip_fp16.h>

typedef __attribute__((ext_vector_type(16))) _Float16 v16h;
typedef __attribute__((ext_vector_type(8)))  float    v8f;

#define GAT_C 128
#define GAT_ED 8
#define NEG_SLOPE 0.2f

// ---------- order-preserving float <-> u32 map (for atomic max on floats) ----
__device__ __forceinline__ unsigned mapf(float f) {
    unsigned u = __float_as_uint(f);
    return (u & 0x80000000u) ? ~u : (u | 0x80000000u);
}
__device__ __forceinline__ float unmapf(unsigned u) {
    u = (u & 0x80000000u) ? (u & 0x7FFFFFFFu) : ~u;
    return __uint_as_float(u);
}

// ---------- init per-node scratch (must re-init every call) -----------------
__global__ void init_ws(unsigned* __restrict__ mmax, float* __restrict__ denom,
                        float* __restrict__ num, int* __restrict__ node_map, int N) {
    int n = blockIdx.x * blockDim.x + threadIdx.x;
    if (n >= N) return;
    mmax[n]     = 0u;     // minimum of the order-preserving map
    denom[n]    = 0.0f;
    num[n]      = 0.0f;
    node_map[n] = -1;
}

// ---------- h = x @ W via v_wmma_f32_16x16x32_f16 ---------------------------
// One wave -> 16 rows. B holds W broadcast across all 16 columns, so every
// column of D is the dot product; lanes with (lane&15)==0 hold M=0..7 (lane 0)
// and M=8..15 (lane 16) in the 8 accumulator VGPRs per the C/D layout.
__global__ void matvec_wmma(const float* __restrict__ x, const float* __restrict__ W,
                            float* __restrict__ h, int N) {
    int lane = threadIdx.x & 31;
    int wave = threadIdx.x >> 5;
    int r0 = (blockIdx.x * (blockDim.x >> 5) + wave) * 16;
    if (r0 >= N) return;                       // wave-uniform: EXEC stays all-1s
    bool hiHalf = lane >= 16;
    int  m   = lane & 15;
    int  row = r0 + m; if (row >= N) row = N - 1;  // clamp tail, keep EXEC full
    const float* xr = x + (long)row * GAT_C;

    v8f c = {};
    #pragma unroll
    for (int kk = 0; kk < GAT_C; kk += 32) {
        v16h a, b;
        #pragma unroll
        for (int i = 0; i < 8; ++i) {
            // 16-bit A-matrix 16x32 layout: lanes 0-15 hold K=0..7,16..23,
            // lanes 16-31 hold K=8..15,24..31 (pairs per VGPR).
            int ko = (i < 4) ? (2 * i) : (16 + 2 * (i - 4));
            if (hiHalf) ko += 8;
            float2 xa = *(const float2*)(xr + kk + ko);
            float2 wb = *(const float2*)(W  + kk + ko);
            a[2 * i]     = (_Float16)xa.x;
            a[2 * i + 1] = (_Float16)xa.y;
            b[2 * i]     = (_Float16)wb.x;     // broadcast W across all N cols
            b[2 * i + 1] = (_Float16)wb.y;
        }
        c = __builtin_amdgcn_wmma_f32_16x16x32_f16(
                false, a, false, b, (short)0, c, false, false);
    }
    if (m == 0) {
        int base = r0 + (hiHalf ? 8 : 0);
        #pragma unroll
        for (int p = 0; p < 8; ++p) {
            int r = base + p;
            if (r < N) h[r] = c[p];
        }
    }
}

// ---------- edge pass 1: e = leaky_relu(a_s[src]+a_d[dst]); segment max -----
__global__ void edge_pass1(const int* __restrict__ ei, const float* __restrict__ h,
                           const float* __restrict__ att_src, const float* __restrict__ att_dst,
                           float* __restrict__ evals, unsigned* __restrict__ mmax, int E) {
    int e = blockIdx.x * blockDim.x + threadIdx.x;
    if (e >= E) return;
    __builtin_prefetch(ei + e + 16384, 0, 0);          // global_prefetch_b8
    int s = ei[e], d = ei[E + e];
    float v = h[s] * att_src[0] + h[d] * att_dst[0];
    v = (v > 0.0f) ? v : v * NEG_SLOPE;
    evals[e] = v;
    atomicMax(mmax + d, mapf(v));
}

// ---------- edge pass 2: softmax weights + fused numerator/denominator ------
__global__ void edge_pass2(const int* __restrict__ ei, const float* __restrict__ h,
                           const float* __restrict__ evals, const unsigned* __restrict__ mmax,
                           float* __restrict__ denom, float* __restrict__ num, int E) {
    int e = blockIdx.x * blockDim.x + threadIdx.x;
    if (e >= E) return;
    __builtin_prefetch(evals + e + 16384, 0, 0);
    int s = ei[e], d = ei[E + e];
    float w = __expf(evals[e] - unmapf(mmax[d]));
    atomicAdd(denom + d, w);
    atomicAdd(num + d, w * h[s]);
}

// ---------- score + 64-bit sort keys (desc score, asc index tie-break) ------
__global__ void score_keys(const float* __restrict__ num, const float* __restrict__ denom,
                           const float* __restrict__ bias,
                           unsigned long long* __restrict__ keys, int N, int M) {
    int n = blockIdx.x * blockDim.x + threadIdx.x;
    if (n >= M) return;
    if (n < N) {
        float dn = denom[n];
        float s  = (dn > 0.0f) ? (num[n] / dn + bias[0]) : -__builtin_inff();
        keys[n] = ((unsigned long long)mapf(s) << 32)
                | (unsigned long long)(0xFFFFFFFFu - (unsigned)n);
    } else {
        keys[n] = 0ull;                         // padding sinks to the end
    }
}

// ---------- bitonic sort step (descending on u64 key), 2MB = L2 resident ----
__global__ void bitonic_step(unsigned long long* __restrict__ keys, int j, int k, int M) {
    int i = blockIdx.x * blockDim.x + threadIdx.x;
    if (i >= M) return;
    int ixj = i ^ j;
    if (ixj > i) {
        unsigned long long a = keys[i], b = keys[ixj];
        bool up = ((i & k) == 0);               // descending overall
        if (up ? (a < b) : (a > b)) { keys[i] = b; keys[ixj] = a; }
    }
}

// ---------- perm, node_map, perm output -------------------------------------
__global__ void build_perm(const unsigned long long* __restrict__ keys,
                           int* __restrict__ node_map, int* __restrict__ perm,
                           float* __restrict__ out_perm, int K) {
    int i = blockIdx.x * blockDim.x + threadIdx.x;
    if (i >= K) return;
    int p = (int)(0xFFFFFFFFu - (unsigned)(keys[i] & 0xFFFFFFFFull));
    perm[i]     = p;
    node_map[p] = i;
    out_perm[i] = (float)p;
}

// ---------- filtered_x = x[perm] : one wave per 128-float row (float4) ------
__global__ void gather_x(const float* __restrict__ x, const int* __restrict__ perm,
                         float* __restrict__ out, int K) {
    int lane = threadIdx.x & 31;
    int row  = blockIdx.x * (blockDim.x >> 5) + (threadIdx.x >> 5);
    if (row >= K) return;
    int p = perm[row];
    const float4* src = (const float4*)(x   + (long)p   * GAT_C);
    float4*       dst = (float4*)      (out + (long)row * GAT_C);
    dst[lane] = src[lane];
}

// ---------- edge remap + mask + attr zeroing --------------------------------
__global__ void edge_filter(const int* __restrict__ ei, const float* __restrict__ attr,
                            const int* __restrict__ node_map,
                            float* __restrict__ out_ei, float* __restrict__ out_attr, int E) {
    int e = blockIdx.x * blockDim.x + threadIdx.x;
    if (e >= E) return;
    __builtin_prefetch(attr + (long)(e + 8192) * GAT_ED, 0, 0);
    int s = ei[e], d = ei[E + e];
    int ms = node_map[s], md = node_map[d];
    bool ok = (ms >= 0) && (md >= 0);
    out_ei[e]     = (float)(ok ? ms : -1);
    out_ei[E + e] = (float)(ok ? md : -1);
    const float4* ap = (const float4*)(attr + (long)e * GAT_ED);
    float4* op = (float4*)(out_attr + (long)e * GAT_ED);
    float4 a0 = ap[0], a1 = ap[1];
    if (!ok) { a0 = make_float4(0.f, 0.f, 0.f, 0.f); a1 = a0; }
    op[0] = a0; op[1] = a1;
}

extern "C" void kernel_launch(void* const* d_in, const int* in_sizes, int n_in,
                              void* d_out, int out_size, void* d_ws, size_t ws_size,
                              hipStream_t stream) {
    const float* x       = (const float*)d_in[0];
    const int*   ei      = (const int*)  d_in[1];
    const float* eattr   = (const float*)d_in[2];
    const float* W       = (const float*)d_in[3];
    const float* att_src = (const float*)d_in[4];
    const float* att_dst = (const float*)d_in[5];
    const float* bias    = (const float*)d_in[6];

    const int N = in_sizes[0] / GAT_C;
    const int E = in_sizes[2] / GAT_ED;
    const int K = (9 * N + 9) / 10;             // ceil(0.9 * N)
    int M = 1; while (M < N) M <<= 1;           // bitonic size (262144 for N=200000)

    // ---- carve workspace ----
    char* ws = (char*)d_ws;
    size_t off = 0;
    auto carve = [&](size_t bytes) -> void* {
        void* p = ws + off;
        off = (off + bytes + 255) & ~(size_t)255;
        return p;
    };
    float*              h        = (float*)             carve((size_t)N * 4);
    unsigned*           mmax     = (unsigned*)          carve((size_t)N * 4);
    float*              denom    = (float*)             carve((size_t)N * 4);
    float*              num      = (float*)             carve((size_t)N * 4);
    int*                node_map = (int*)               carve((size_t)N * 4);
    int*                perm     = (int*)               carve((size_t)K * 4);
    unsigned long long* keys     = (unsigned long long*)carve((size_t)M * 8);
    float*              evals    = (float*)             carve((size_t)E * 4);

    // ---- output layout (concatenated flat, reference return order) ----
    float* out_x    = (float*)d_out;                       // K*128
    float* out_ei   = out_x    + (size_t)K * GAT_C;        // 2*E
    float* out_attr = out_ei   + (size_t)2 * E;            // E*8
    float* out_perm = out_attr + (size_t)E * GAT_ED;       // K

    const int B = 256;
    // 1) init L2-resident per-node scratch
    init_ws<<<(N + B - 1) / B, B, 0, stream>>>(mmax, denom, num, node_map, N);
    // 2) h = x @ W  (WMMA, 16 rows per wave, 8 waves per block)
    matvec_wmma<<<(N + 127) / 128, B, 0, stream>>>(x, W, h, N);
    // 3) attention logits + segment max
    edge_pass1<<<(E + B - 1) / B, B, 0, stream>>>(ei, h, att_src, att_dst, evals, mmax, E);
    // 4) softmax weights + fused numerator/denominator
    edge_pass2<<<(E + B - 1) / B, B, 0, stream>>>(ei, h, evals, mmax, denom, num, E);
    // 5) scores -> packed descending sort keys (+ padding)
    score_keys<<<(M + B - 1) / B, B, 0, stream>>>(num, denom, bias, keys, N, M);
    // 6) full bitonic sort (descending); 2MB working set stays in L2
    for (int k = 2; k <= M; k <<= 1)
        for (int j = k >> 1; j > 0; j >>= 1)
            bitonic_step<<<(M + B - 1) / B, B, 0, stream>>>(keys, j, k, M);
    // 7) perm / node_map / perm output
    build_perm<<<(K + B - 1) / B, B, 0, stream>>>(keys, node_map, perm, out_perm, K);
    // 8) filtered_x gather (wave per row, float4)
    gather_x<<<(K + 7) / 8, B, 0, stream>>>(x, perm, out_x, K);
    // 9) edge remap + mask + attr zeroing
    edge_filter<<<(E + B - 1) / B, B, 0, stream>>>(ei, eattr, node_map, out_ei, out_attr, E);
}
